// BetweennessRoPE_343597383767
// MI455X (gfx1250) — compile-verified, matrix operand order checked by reading, not run on hardware
//
#include <hip/hip_runtime.h>
#include <math.h>

// ---------------------------------------------------------------------------
// BetweennessRoPE for MI455X (gfx1250, wave32, WMMA).
//
// Problem sizes (fixed by the reference): B=4, S=2048, H=16, D=128.
// content = x @ W^T (+b) is only used via differences of neighboring rows,
// so the bias cancels exactly and is never loaded. The score perturbs the
// RoPE position by <5e-5 of a step, so the projection is done in f16 WMMA
// with f32 accumulation (v_wmma_f32_16x16x32_f16) — numerically free,
// and the kernel becomes purely HBM-bandwidth bound (~200 MiB total traffic
// => ~9 us at 23.3 TB/s).
// ---------------------------------------------------------------------------

#define B_N   4
#define S_LEN 2048
#define H_N   16
#define D_DIM 128
#define N_SEQ (B_N * H_N)          // 64 independent sequences
#define ROWS  32                   // content rows per block (2 x 16 M-tiles)
#define TS_OUT 30                  // scores produced per block (ROWS - 2 halo)
#define ADJ_SCALE 0.1f
#define LOG2_10000 13.287712379549449f

typedef _Float16 h16 __attribute__((ext_vector_type(16)));
typedef _Float16 h8  __attribute__((ext_vector_type(8)));
typedef float    vf8 __attribute__((ext_vector_type(8)));

// ---------------------------------------------------------------------------
// Kernel A: f16 WMMA projection of a 32-token tile, then d01/d02 neighbor
// distances and the betweenness score  ->  ws_between[n*S + s].
// grid = (ceil((S-2)/TS_OUT)=69, N_SEQ=64), block = 256 (8 waves).
// ---------------------------------------------------------------------------
__global__ __launch_bounds__(256)
void betweenness_score_kernel(const float* __restrict__ x,
                              const float* __restrict__ W,
                              float* __restrict__ ws_between)
{
    __shared__ _Float16 Alds[ROWS][D_DIM];        // x tile, f16, row-major
    __shared__ _Float16 Wlds[D_DIM][D_DIM];       // W,     f16, row-major (e,d)
    __shared__ float    Clds[ROWS][D_DIM + 1];    // content tile (+1: bank pad)
    __shared__ float    d01l[ROWS - 1];
    __shared__ float    d02l[ROWS - 2];

    const int n   = blockIdx.y;                   // sequence id = b*H + h
    const int b   = n >> 4;
    const int h   = n & 15;
    const int c0  = blockIdx.x * TS_OUT;          // first content row (global s)
    const int tid = threadIdx.x;

    // ---- load x tile (f32 -> f16), rows clamped at S-1 for the tail block
#pragma unroll
    for (int i = 0; i < (ROWS * D_DIM) / 256; ++i) {
        int idx = tid + i * 256;
        int r = idx >> 7, c = idx & 127;
        int s = c0 + r; if (s > S_LEN - 1) s = S_LEN - 1;
        Alds[r][c] = (_Float16)x[((b * S_LEN + s) * H_N + h) * D_DIM + c];
    }
    // ---- load W (f32 -> f16); W row e is B-matrix column e (K-contiguous)
#pragma unroll
    for (int i = 0; i < (D_DIM * D_DIM) / 256; ++i) {
        int idx = tid + i * 256;
        Wlds[idx >> 7][idx & 127] = (_Float16)W[idx];
    }
    __syncthreads();

    // ---- WMMA: 2 M-tiles x 8 N-tiles = 16 jobs over 8 waves (2 each).
    // A-fragment lane layout (16-bit A 16x32): lane<16 -> M=lane,
    //   halves 0..7 = K[k0..k0+7], halves 8..15 = K[k0+16..k0+23];
    // lane>=16 -> same M, K shifted by +8. B mirrors with lane = N column.
    const int wave = tid >> 5;
    const int lane = tid & 31;
    const int lmod = lane & 15;
    const int lsel = lane >> 4;

#pragma unroll
    for (int jj = 0; jj < 2; ++jj) {
        const int job = wave * 2 + jj;
        const int mt = job >> 3;                  // 0..1  (token tile)
        const int nt = job & 7;                   // 0..7  (channel tile)
        vf8 acc = {};
#pragma unroll
        for (int k0 = 0; k0 < D_DIM; k0 += 32) {
            const h8* ap = (const h8*)&Alds[mt * 16 + lmod][k0 + lsel * 8];
            h8 alo = ap[0];
            h8 ahi = ap[2];                       // +16 halves
            h16 a = __builtin_shufflevector(alo, ahi,
                     0,1,2,3,4,5,6,7,8,9,10,11,12,13,14,15);
            const h8* bp = (const h8*)&Wlds[nt * 16 + lmod][k0 + lsel * 8];
            h8 blo = bp[0];
            h8 bhi = bp[2];
            h16 bv = __builtin_shufflevector(blo, bhi,
                     0,1,2,3,4,5,6,7,8,9,10,11,12,13,14,15);
            acc = __builtin_amdgcn_wmma_f32_16x16x32_f16(
                      false, a, false, bv, (short)0, acc, false, false);
        }
        // C/D layout: VGPR j -> row (mt*16 + j + 8*lsel), col (nt*16 + lmod)
#pragma unroll
        for (int j = 0; j < 8; ++j)
            Clds[mt * 16 + j + lsel * 8][nt * 16 + lmod] = acc[j];
    }
    __syncthreads();

    // ---- neighbor distances over the channel dim (bias cancels here)
    if (tid < ROWS - 1) {
        float s2 = 0.0f;
#pragma unroll 8
        for (int e = 0; e < D_DIM; ++e) {
            float d = Clds[tid][e] - Clds[tid + 1][e];
            s2 = fmaf(d, d, s2);
        }
        d01l[tid] = sqrtf(s2);
    } else if (tid >= 64 && tid < 64 + ROWS - 2) {
        const int t = tid - 64;
        float s2 = 0.0f;
#pragma unroll 8
        for (int e = 0; e < D_DIM; ++e) {
            float d = Clds[t][e] - Clds[t + 2][e];
            s2 = fmaf(d, d, s2);
        }
        d02l[t] = sqrtf(s2);
    }
    __syncthreads();

    // ---- score[s] needs d01[s-1], d01[s], d02[s-1]; local idx of s is tid+1
    if (tid < TS_OUT) {
        const int s = c0 + 1 + tid;
        if (s <= S_LEN - 2) {
            const float direct = d02l[tid];
            const float path   = d01l[tid] + d01l[tid + 1];
            float sc = 1.0f - (path - direct) / fmaxf(direct, 1e-6f);
            sc = fmaxf(sc, 0.0f);
            ws_between[n * S_LEN + s] = sc * (1.0f / (float)(S_LEN - 2));
        }
    }
}

// ---------------------------------------------------------------------------
// Kernel B: streaming RoPE with interpolated cos/sin tables, one float2
// (even/odd channel pair) per thread, fully coalesced.
// ---------------------------------------------------------------------------
__global__ __launch_bounds__(256)
void betweenness_rope_kernel(const float* __restrict__ x,
                             const float* __restrict__ ws_between,
                             float* __restrict__ out)
{
    const int g = blockIdx.x * 256 + threadIdx.x;     // pair index
    const int j = g & 63;                             // pair 0..63
    const int h = (g >> 6) & 15;
    const int s = (g >> 10) & (S_LEN - 1);
    const int b = g >> 21;

    float bet = 0.0f;
    if (s >= 1 && s <= S_LEN - 2)
        bet = ws_between[(b * H_N + h) * S_LEN + s];

    float pos = (float)s + (bet - 0.5f) * ADJ_SCALE;
    pos = fminf(fmaxf(pos, 0.0f), (float)(S_LEN - 1));
    const float lo   = floorf(pos);
    const float hi   = ceilf(pos);
    const float frac = pos - lo;

    // base_j = 10000^(-(2j)/D)
    const float base = exp2f((float)(2 * j) * (-LOG2_10000 / (float)D_DIM));
    float slo, clo, shi, chi;
    __sincosf(lo * base, &slo, &clo);
    __sincosf(hi * base, &shi, &chi);
    const float ci = (1.0f - frac) * clo + frac * chi;
    const float si = (1.0f - frac) * slo + frac * shi;

    const int idx = ((b * S_LEN + s) * H_N + h) * (D_DIM / 2) + j;
    const float2 v = ((const float2*)x)[idx];
    float2 o;
    o.x = v.x * ci - v.y * si;   // oe = xe*cos - xo*sin
    o.y = v.y * ci + v.x * si;   // oo = xo*cos + xe*sin
    ((float2*)out)[idx] = o;
}

// ---------------------------------------------------------------------------
extern "C" void kernel_launch(void* const* d_in, const int* in_sizes, int n_in,
                              void* d_out, int out_size, void* d_ws, size_t ws_size,
                              hipStream_t stream)
{
    const float* x = (const float*)d_in[0];
    const float* W = (const float*)d_in[1];
    // d_in[2] = bias b: unused — it cancels in all distance computations and
    // content is not otherwise observed.
    float* between = (float*)d_ws;           // N_SEQ * S_LEN floats = 512 KB
    float* out     = (float*)d_out;

    dim3 gridA((S_LEN - 2 + TS_OUT - 1) / TS_OUT, N_SEQ);     // 69 x 64
    betweenness_score_kernel<<<gridA, 256, 0, stream>>>(x, W, between);

    const int total_pairs = B_N * S_LEN * H_N * (D_DIM / 2);  // 8,388,608
    betweenness_rope_kernel<<<dim3(total_pairs / 256), 256, 0, stream>>>(
        x, between, out);
}
